// CONVSPEmbedder_21062519620287
// MI455X (gfx1250) — matrix operand order repaired and checked by reading, not compiled, not run on hardware
//
#include <hip/hip_runtime.h>
#include <hip/hip_bf16.h>

// Problem constants (match reference)
#define NN   102400
#define BB   1024
#define NPG  100
#define EPG  1600
#define EE   (BB*EPG)
#define DD   512
#define LL   3
#define EPSV 1e-5f
#define SLOPEV 0.1f

typedef __attribute__((ext_vector_type(16))) _Float16 v16h;
typedef __attribute__((ext_vector_type(8)))  _Float16 v8h;
typedef __attribute__((ext_vector_type(8)))  float    v8f;

__device__ __forceinline__ float leaky(float x) { return x > 0.f ? x : SLOPEV * x; }

// ---------------- utility ----------------
__global__ void k_zero(float* __restrict__ p, int n) {
    int i = blockIdx.x * blockDim.x + threadIdx.x;
    if (i < n) p[i] = 0.f;
}

// ---------------- degrees ----------------
__global__ void k_degree(const int* __restrict__ src, const int* __restrict__ dst,
                         float* __restrict__ dout, float* __restrict__ din) {
    int e = blockIdx.x * blockDim.x + threadIdx.x;
    if (e < EE) {
        atomicAdd(&dout[src[e]], 1.f);
        atomicAdd(&din[dst[e]], 1.f);
    }
}

__global__ void k_cfac(float* __restrict__ dout, float* __restrict__ din) {
    int i = blockIdx.x * blockDim.x + threadIdx.x;
    if (i < NN) {
        dout[i] = rsqrtf(fmaxf(dout[i], 1.f));   // c_out
        din[i]  = rsqrtf(fmaxf(din[i], 1.f));    // c_in
    }
}

// ---------------- per-graph edge aggregation in LDS ----------------
// One workgroup per graph: stage h*c_out and msg accumulator in LDS
// (feature dim split into 2 chunks of 256 so both fit in 320KB WGP LDS),
// serialize edges, parallelize features across the 256 lanes.
// Output: A16 = (msg * c_in) as f16, the GEMM A operand.
__global__ void __launch_bounds__(256)
k_aggregate(const float* __restrict__ hin, const float* __restrict__ cof,
            const float* __restrict__ cif, const int* __restrict__ src,
            const int* __restrict__ dst, _Float16* __restrict__ A16) {
    const int g = blockIdx.x;
    const int t = threadIdx.x;
    extern __shared__ char smem[];
    int*   es   = (int*)smem;                               // [2*EPG]
    float* hbuf = (float*)(smem + 2 * EPG * sizeof(int));   // [NPG*256]
    float* mbuf = hbuf + NPG * 256;                         // [NPG*256]
    const int nbase = g * NPG;
    const int ebase = g * EPG;
    for (int e = t; e < EPG; e += 256) {
        es[2 * e]     = src[ebase + e] - nbase;
        es[2 * e + 1] = dst[ebase + e] - nbase;
    }
    for (int c = 0; c < 2; ++c) {
        __syncthreads();
        for (int i = 0; i < NPG; ++i) {
            hbuf[i * 256 + t] = hin[(size_t)(nbase + i) * DD + c * 256 + t] * cof[nbase + i];
            mbuf[i * 256 + t] = 0.f;
        }
        __syncthreads();
#pragma unroll 8
        for (int e = 0; e < EPG; ++e) {
            int s = es[2 * e], d = es[2 * e + 1];
            mbuf[d * 256 + t] += hbuf[s * 256 + t];
        }
        __syncthreads();
        for (int i = 0; i < NPG; ++i) {
            float v = mbuf[i * 256 + t] * cif[nbase + i];
            A16[(size_t)(nbase + i) * DD + c * 256 + t] = (_Float16)v;
        }
    }
}

// ---------------- W convert: f32 row-major -> f16 column-major ----------------
// Column-major W makes the WMMA B-fragment (K contiguous per lane) a pair of
// contiguous b128 LDS loads.
__global__ void k_convW(const float* __restrict__ W, _Float16* __restrict__ Wt) {
    int idx = blockIdx.x * blockDim.x + threadIdx.x;
    if (idx < DD * DD) {
        int n = idx >> 9, k = idx & 511;
        Wt[n * DD + k] = (_Float16)W[k * DD + n];
    }
}

// ---------------- WMMA GEMM: h = A16[N,512] @ W[512,512] ----------------
// 256 threads = 8 waves. Workgroup tile 128x128, K step 32, wave tile 32x64
// (2x4 grid of 16x16 accumulators). Double-buffered LDS: next tile's global
// loads issue before the barrier and land in registers while the current
// tile's 8 WMMAs execute; one barrier per K-step. Fragment layouts per the
// CDNA5 ISA tables.
#define LDA 40   // padded LDS row stride (f16 elems); multiple of 8, bank-spread
__global__ void __launch_bounds__(256)
k_gemm(const _Float16* __restrict__ A, const _Float16* __restrict__ Wt,
       float* __restrict__ C) {
    __shared__ __align__(16) _Float16 As[2][128 * LDA];   // [buf][row][k]
    __shared__ __align__(16) _Float16 Bs[2][128 * LDA];   // [buf][col][k] (B^T)
    const int t    = threadIdx.x;
    const int lane = t & 31;
    const int wave = t >> 5;
    const int wm   = (wave & 3) * 32;   // wave row offset in tile
    const int wn   = (wave >> 2) * 64;  // wave col offset in tile
    const int bm   = (blockIdx.x >> 2) * 128;
    const int bn   = (blockIdx.x & 3) * 128;
    const int r    = t >> 1;            // 0..127: tile row (A) / tile col (B)
    const int seg  = (t & 1) * 16;      // which 16-elem half of the 32-K row

    const size_t arow = (size_t)(bm + r) * DD + seg;   // A stream row base
    const size_t brow = (size_t)(bn + r) * DD + seg;   // W^T col base

    v8f zero = {};
    v8f acc[2][4];
#pragma unroll
    for (int i = 0; i < 2; ++i)
#pragma unroll
        for (int j = 0; j < 4; ++j) acc[i][j] = zero;

    const int half = lane >> 4;   // 0: lanes 0-15, 1: lanes 16-31
    const int mrow = lane & 15;

    // prologue: stage tile 0 into buffer 0
    {
        v8h a0 = *(const v8h*)&A[arow];
        v8h a1 = *(const v8h*)&A[arow + 8];
        v8h b0 = *(const v8h*)&Wt[brow];
        v8h b1 = *(const v8h*)&Wt[brow + 8];
        *(v8h*)&As[0][r * LDA + seg]     = a0;
        *(v8h*)&As[0][r * LDA + seg + 8] = a1;
        *(v8h*)&Bs[0][r * LDA + seg]     = b0;
        *(v8h*)&Bs[0][r * LDA + seg + 8] = b1;
    }

    for (int kt = 0; kt < 16; ++kt) {
        const int cur = kt & 1;
        // issue next tile's global loads before the barrier (latency hidden
        // behind this tile's WMMAs); prefetch the A stream two tiles ahead.
        v8h na0, na1, nb0, nb1;
        if (kt < 15) {
            const int k1 = (kt + 1) * 32;
            na0 = *(const v8h*)&A[arow + k1];
            na1 = *(const v8h*)&A[arow + k1 + 8];
            nb0 = *(const v8h*)&Wt[brow + k1];
            nb1 = *(const v8h*)&Wt[brow + k1 + 8];
            if (kt < 14)
                __builtin_prefetch(&A[arow + (kt + 2) * 32], 0, 1);
        }
        __syncthreads();   // buffer `cur` fully staged for all waves

        // A fragments: 16x32 f16, lane (m = lane&15, half):
        //   K(j) = (j>=8?16:0) + half*8 + (j&7)  -> two contiguous b128 loads
        v16h a[2];
#pragma unroll
        for (int i = 0; i < 2; ++i) {
            const _Float16* p = &As[cur][(wm + i * 16 + mrow) * LDA];
            v8h lo = *(const v8h*)&p[half * 8];
            v8h hi = *(const v8h*)&p[16 + half * 8];
#pragma unroll
            for (int j = 0; j < 8; ++j) { a[i][j] = lo[j]; a[i][8 + j] = hi[j]; }
        }
        // B fragments: 32x16 f16, lane (n = lane&15, half): K(j) = (j>=8?16:0) + half*8 + (j&7)
        v16h b[4];
#pragma unroll
        for (int j4 = 0; j4 < 4; ++j4) {
            const _Float16* p = &Bs[cur][(wn + j4 * 16 + mrow) * LDA];
            v8h lo = *(const v8h*)&p[half * 8];
            v8h hi = *(const v8h*)&p[16 + half * 8];
#pragma unroll
            for (int j = 0; j < 8; ++j) { b[j4][j] = lo[j]; b[j4][8 + j] = hi[j]; }
        }
#pragma unroll
        for (int i = 0; i < 2; ++i)
#pragma unroll
            for (int j = 0; j < 4; ++j)
                acc[i][j] = __builtin_amdgcn_wmma_f32_16x16x32_f16(
                    false, a[i], false, b[j], (short)0, acc[i][j], false, false);

        // stage next tile into the alternate buffer (readers of `cur` are
        // unaffected; next iteration's barrier publishes it)
        if (kt < 15) {
            const int nxt = cur ^ 1;
            *(v8h*)&As[nxt][r * LDA + seg]     = na0;
            *(v8h*)&As[nxt][r * LDA + seg + 8] = na1;
            *(v8h*)&Bs[nxt][r * LDA + seg]     = nb0;
            *(v8h*)&Bs[nxt][r * LDA + seg + 8] = nb1;
        }
    }

    // C/D layout: VGPR v -> row = v + 8*(lane>=16); col = lane&15
#pragma unroll
    for (int i = 0; i < 2; ++i) {
        const int rbase = bm + wm + i * 16 + half * 8;
#pragma unroll
        for (int j = 0; j < 4; ++j) {
            const int col = bn + wn + j * 16 + mrow;
#pragma unroll
            for (int v = 0; v < 8; ++v)
                C[(size_t)(rbase + v) * DD + col] = acc[i][j][v];
        }
    }
}

// ---------------- batch statistics (sum / sumsq per feature) ----------------
__global__ void k_bstat(const float* __restrict__ h, float* __restrict__ bsum,
                        float* __restrict__ bss) {
    const int fb = blockIdx.x & 1;
    const int rb = blockIdx.x >> 1;        // 0..199, each covers 512 rows
    const int f  = fb * 256 + threadIdx.x;
    const int r0 = rb * 512;
    float s = 0.f, ss = 0.f;
    for (int r = 0; r < 512; ++r) {
        float x = h[(size_t)(r0 + r) * DD + f];
        s += x; ss += x * x;
    }
    atomicAdd(&bsum[f], s);
    atomicAdd(&bss[f], ss);
}

__global__ void k_bfin(const float* __restrict__ bsum, const float* __restrict__ bss,
                       float* __restrict__ bmu, float* __restrict__ brs) {
    int f = blockIdx.x * 256 + threadIdx.x;
    if (f < DD) {
        float mu  = bsum[f] * (1.f / NN);
        float var = bss[f] * (1.f / NN) - mu * mu;
        bmu[f] = mu;
        brs[f] = rsqrtf(var + EPSV);
    }
}

// ---------------- fused UnitedNorm + leaky + graph-mean readout ----------------
// One workgroup per graph. Graph stats + node stats computed in-block,
// mix applied in place on h, readout slice written with final leaky.
__global__ void __launch_bounds__(256)
k_norm(float* __restrict__ h, const float* __restrict__ bmu,
       const float* __restrict__ brs, const float* __restrict__ gam,
       const float* __restrict__ bet, const float* __restrict__ lam,
       float* __restrict__ out, int layer) {
    __shared__ float smg[DD], srg[DD], smn[NPG], srn[NPG];
    const int g = blockIdx.x, t = threadIdx.x;
    const size_t base = (size_t)g * NPG * DD;

    // graph stats for features t and t+256
    {
        float s0 = 0, ss0 = 0, s1 = 0, ss1 = 0;
        for (int i = 0; i < NPG; ++i) {
            float x0 = h[base + (size_t)i * DD + t];
            float x1 = h[base + (size_t)i * DD + t + 256];
            s0 += x0; ss0 += x0 * x0;
            s1 += x1; ss1 += x1 * x1;
        }
        float mu0 = s0 * (1.f / NPG), mu1 = s1 * (1.f / NPG);
        smg[t]       = mu0; srg[t]       = rsqrtf(ss0 * (1.f / NPG) - mu0 * mu0 + EPSV);
        smg[t + 256] = mu1; srg[t + 256] = rsqrtf(ss1 * (1.f / NPG) - mu1 * mu1 + EPSV);
    }
    // node stats: one node per thread (first 100 threads)
    if (t < NPG) {
        float s = 0, ss = 0;
        for (int f = 0; f < DD; ++f) {
            float x = h[base + (size_t)t * DD + f];
            s += x; ss += x * x;
        }
        float mu = s * (1.f / DD);
        smn[t] = mu;
        srn[t] = rsqrtf(ss * (1.f / DD) - mu * mu + EPSV);
    }
    __syncthreads();

    const float l0 = lam[0], l1 = lam[1], l2 = lam[2];
    const float g0 = gam[t], b0 = bet[t], g1 = gam[t + 256], b1 = bet[t + 256];
    const float bm0 = bmu[t], br0 = brs[t], bm1 = bmu[t + 256], br1 = brs[t + 256];
    const float gm0 = smg[t], gr0 = srg[t], gm1 = smg[t + 256], gr1 = srg[t + 256];
    float acc0 = 0.f, acc1 = 0.f;
    for (int i = 0; i < NPG; ++i) {
        const float mn = smn[i], rn = srn[i];
        float x0 = h[base + (size_t)i * DD + t];
        float y0 = g0 * (l0 * (x0 - bm0) * br0 + l1 * (x0 - gm0) * gr0 + l2 * (x0 - mn) * rn) + b0;
        y0 = leaky(y0);
        h[base + (size_t)i * DD + t] = y0;
        acc0 += y0;
        float x1 = h[base + (size_t)i * DD + t + 256];
        float y1 = g1 * (l0 * (x1 - bm1) * br1 + l1 * (x1 - gm1) * gr1 + l2 * (x1 - mn) * rn) + b1;
        y1 = leaky(y1);
        h[base + (size_t)i * DD + t + 256] = y1;
        acc1 += y1;
    }
    out[(size_t)g * (3 * DD) + layer * DD + t]       = leaky(acc0 * (1.f / NPG));
    out[(size_t)g * (3 * DD) + layer * DD + t + 256] = leaky(acc1 * (1.f / NPG));
}

// ---------------- host-side launcher ----------------
extern "C" void kernel_launch(void* const* d_in, const int* in_sizes, int n_in,
                              void* d_out, int out_size, void* d_ws, size_t ws_size,
                              hipStream_t stream) {
    (void)in_sizes; (void)n_in; (void)out_size; (void)ws_size;
    const float* node_feats = (const float*)d_in[0];
    const int*   src        = (const int*)d_in[1];
    const int*   dst        = (const int*)d_in[2];
    /* d_in[3] = graph_ids (unused: block structure is implicit) */
    const float* W          = (const float*)d_in[4];
    const float* gam        = (const float*)d_in[5];
    const float* bet        = (const float*)d_in[6];
    const float* lambdas    = (const float*)d_in[7];
    float* out = (float*)d_out;

    char* ws = (char*)d_ws;
    size_t off = 0;
    auto alloc = [&](size_t bytes) -> char* {
        char* p = ws + off;
        off += (bytes + 255) & ~(size_t)255;
        return p;
    };
    float*    c_out = (float*)alloc((size_t)NN * 4);
    float*    c_in  = (float*)alloc((size_t)NN * 4);
    float*    hbuf  = (float*)alloc((size_t)NN * DD * 4);
    _Float16* A16   = (_Float16*)alloc((size_t)NN * DD * 2);
    _Float16* W16   = (_Float16*)alloc((size_t)DD * DD * 2);
    float*    bsum  = (float*)alloc(DD * 4);
    float*    bss   = (float*)alloc(DD * 4);
    float*    bmu   = (float*)alloc(DD * 4);
    float*    brs   = (float*)alloc(DD * 4);

    // degrees -> normalization factors
    k_zero<<<(NN + 255) / 256, 256, 0, stream>>>(c_out, NN);
    k_zero<<<(NN + 255) / 256, 256, 0, stream>>>(c_in, NN);
    k_degree<<<EE / 256, 256, 0, stream>>>(src, dst, c_out, c_in);
    k_cfac<<<NN / 256, 256, 0, stream>>>(c_out, c_in);

    const size_t aggsh = 2 * EPG * sizeof(int) + 2 * (size_t)NPG * 256 * sizeof(float);
    const float* hin = node_feats;
    for (int l = 0; l < LL; ++l) {
        k_aggregate<<<BB, 256, aggsh, stream>>>(hin, c_out, c_in, src, dst, A16);
        k_convW<<<DD * DD / 256, 256, 0, stream>>>(W + (size_t)l * DD * DD, W16);
        k_gemm<<<(NN / 128) * (DD / 128), 256, 0, stream>>>(A16, W16, hbuf);
        k_zero<<<2, 256, 0, stream>>>(bsum, DD);
        k_zero<<<2, 256, 0, stream>>>(bss, DD);
        k_bstat<<<400, 256, 0, stream>>>(hbuf, bsum, bss);
        k_bfin<<<2, 256, 0, stream>>>(bsum, bss, bmu, brs);
        k_norm<<<BB, 256, 0, stream>>>(hbuf, bmu, brs, gam + (size_t)l * DD,
                                       bet + (size_t)l * DD, lambdas + (size_t)l * 3,
                                       out, l);
        hin = hbuf;
    }
}
